// CGRU_74139725464080
// MI455X (gfx1250) — compile-verified, matrix operand order checked by reading, not run on hardware
//
#include <hip/hip_runtime.h>
#include <hip/hip_bf16.h>
#include <stdint.h>

// ---------------------------------------------------------------------------
// CDNA5 / gfx1250 implementation of the spectral-norm GRU reference.
// All heavy GEMMs run through v_wmma_f32_16x16x32_bf16 (fp32 accumulate).
// ---------------------------------------------------------------------------

typedef __attribute__((ext_vector_type(16))) __bf16 v16bf;
typedef __attribute__((ext_vector_type(8)))  __bf16 v8bf;
typedef __attribute__((ext_vector_type(8)))  float  v8f;

static constexpr int Bsz = 256;     // batch
static constexpr int Zd  = 512;     // z / x dim
static constexpr int Hd  = 2048;    // hidden dim
static constexpr int Td  = 16;      // timesteps
static constexpr int G3  = 3 * Hd;  // 6144

// ---------------------------------------------------------------------------
// fp32 -> bf16 conversion (grid-stride)
// ---------------------------------------------------------------------------
__global__ void k_f32_to_bf16(const float* __restrict__ src,
                              __bf16* __restrict__ dst, int n) {
  int i = blockIdx.x * blockDim.x + threadIdx.x;
  int stride = gridDim.x * blockDim.x;
  for (; i < n; i += stride) dst[i] = (__bf16)src[i];
}

__global__ void k_zero_u32(uint32_t* __restrict__ p, int n) {
  int i = blockIdx.x * blockDim.x + threadIdx.x;
  int stride = gridDim.x * blockDim.x;
  for (; i < n; i += stride) p[i] = 0u;
}

// ---------------------------------------------------------------------------
// Spectral norm: t = W^T u ; v = t/(|t|+eps) ; wv = W v ; 1/sigma
// W is 512x512 fp32; tiny (<0.3% of total FLOPs) -> simple kernels.
// ---------------------------------------------------------------------------
__global__ void k_wT_u(const float* __restrict__ W, const float* __restrict__ u,
                       float* __restrict__ t) {
  int j = blockIdx.x * 256 + threadIdx.x;
  if (j >= Zd) return;
  float acc = 0.f;
  for (int i = 0; i < Zd; ++i) acc += W[(size_t)i * Zd + j] * u[i];
  t[j] = acc;
}

__global__ void k_normalize(const float* __restrict__ t, float* __restrict__ v) {
  __shared__ float red[256];
  float a = t[threadIdx.x], b = t[threadIdx.x + 256];
  red[threadIdx.x] = a * a + b * b;
  __syncthreads();
  for (int s = 128; s > 0; s >>= 1) {
    if (threadIdx.x < s) red[threadIdx.x] += red[threadIdx.x + s];
    __syncthreads();
  }
  float inv = 1.f / (sqrtf(red[0]) + 1e-12f);
  v[threadIdx.x]       = a * inv;
  v[threadIdx.x + 256] = b * inv;
}

__global__ void k_w_v(const float* __restrict__ W, const float* __restrict__ v,
                      float* __restrict__ wv) {
  int i = blockIdx.x * 256 + threadIdx.x;
  if (i >= Zd) return;
  const float* row = W + (size_t)i * Zd;
  float acc = 0.f;
  for (int j = 0; j < Zd; ++j) acc += row[j] * v[j];
  wv[i] = acc;
}

__global__ void k_sigma(const float* __restrict__ wv, float* __restrict__ scal) {
  __shared__ float red[256];
  float a = wv[threadIdx.x], b = wv[threadIdx.x + 256];
  red[threadIdx.x] = a * a + b * b;
  __syncthreads();
  for (int s = 128; s > 0; s >>= 1) {
    if (threadIdx.x < s) red[threadIdx.x] += red[threadIdx.x + s];
    __syncthreads();
  }
  if (threadIdx.x == 0) {
    float ss = red[0];
    // sigma = u1.(Wv) = ss / (|Wv| + eps)
    float sigma = ss / (sqrtf(ss) + 1e-12f);
    scal[0] = 1.f / sigma;
  }
}

// ---------------------------------------------------------------------------
// WMMA bf16 GEMM:  C[M,N] = (A[M,K] . W[N,K]^T) * scale + bias[N]
//   A row-major bf16 (lda = K elems), W row-major bf16 (ldw = K elems).
//   fp32 output (ldo) and/or bf16 output (ldob) — either may be null.
//
// Block = 128 threads (4 waves in 2x2). Wave tile = 32 x (16*WN):
//   2 A-frags x WN B-frags -> 2*WN wmma per 32-K chunk.
// Block tile: BM=64, BN=32*WN. K chunked by 32, double-buffered LDS
// (pad stride 40 bf16 = 80B: 16B-aligned, conflict-free for 16-row groups).
//
// Fragment layout per ISA 7.12.2 (16-bit A 16x32): lane L<16 holds row M=L,
// K = {kh..kh+7, kh+16..kh+23} with kh=0; lanes 16-31 the same rows with
// kh=8. B is loaded column-symmetric (lane = N column, same K split).
// ---------------------------------------------------------------------------
template <int WN>   // B-frags per wave: 2 (BN=64) or 4 (BN=128)
__global__ __launch_bounds__(128)
void k_gemm_bf16(const __bf16* __restrict__ A, int lda,
                 const __bf16* __restrict__ W, int ldw,
                 const float* __restrict__ bias,
                 const float* __restrict__ scale,      // nullable (1/sigma)
                 float* __restrict__ outF, int ldo,    // nullable
                 __bf16* __restrict__ outB, int ldob,  // nullable
                 int K) {
  constexpr int BM  = 64;
  constexpr int BN  = 32 * WN;      // 2 waves in n, 16*WN cols each
  constexpr int PAD = 40;           // bf16 elems per LDS row (80B)
  constexpr int WSL = BN / 32;      // W-tile row slices per thread (2 or 4)
  __shared__ alignas(16) __bf16 sA[2][BM * PAD];
  __shared__ alignas(16) __bf16 sW[2][BN * PAD];

  const int tid  = threadIdx.x;
  const int lane = tid & 31;
  const int wave = tid >> 5;
  const int wm   = wave >> 1;          // 0..1
  const int wn   = wave & 1;           // 0..1
  const int m0   = blockIdx.y * BM;
  const int n0   = blockIdx.x * BN;

  // Cooperative load mapping: 128 threads = 32 rows x 4 segments of 8 bf16.
  const int lrow = tid >> 2;           // 0..31
  const int lseg = (tid & 3) * 8;      // 0,8,16,24
  const __bf16* pA = A + (size_t)(m0 + lrow) * lda + lseg;
  const __bf16* pW = W + (size_t)(n0 + lrow) * ldw + lseg;

  const int nk = K >> 5;               // K / 32

  // prefetch chunk 0 into registers, then LDS buffer 0
  v8bf ra[2], rw[WSL];
#pragma unroll
  for (int i = 0; i < 2; ++i)   ra[i] = *(const v8bf*)(pA + (size_t)(32 * i) * lda);
#pragma unroll
  for (int i = 0; i < WSL; ++i) rw[i] = *(const v8bf*)(pW + (size_t)(32 * i) * ldw);
#pragma unroll
  for (int i = 0; i < 2; ++i)
    *(v8bf*)&sA[0][(lrow + 32 * i) * PAD + lseg] = ra[i];
#pragma unroll
  for (int i = 0; i < WSL; ++i)
    *(v8bf*)&sW[0][(lrow + 32 * i) * PAD + lseg] = rw[i];
  __syncthreads();

  v8f c[2][WN];
#pragma unroll
  for (int mi = 0; mi < 2; ++mi)
#pragma unroll
    for (int ni = 0; ni < WN; ++ni) c[mi][ni] = (v8f){};

  const int fr = lane & 15;            // fragment row (A) / column (B)
  const int kh = (lane >> 4) * 8;      // K half offset per lane group

  for (int kb = 0; kb < nk; ++kb) {
    const int cur = kb & 1;
    if (kb + 1 < nk) {                 // register prefetch of next chunk
      const __bf16* qA = pA + (size_t)(kb + 1) * 32;
      const __bf16* qW = pW + (size_t)(kb + 1) * 32;
#pragma unroll
      for (int i = 0; i < 2; ++i)   ra[i] = *(const v8bf*)(qA + (size_t)(32 * i) * lda);
#pragma unroll
      for (int i = 0; i < WSL; ++i) rw[i] = *(const v8bf*)(qW + (size_t)(32 * i) * ldw);
    }
    if (kb + 2 < nk) {                 // L2->L0 prefetch two chunks ahead
      __builtin_prefetch(pA + (size_t)(kb + 2) * 32, 0, 3);
      __builtin_prefetch(pW + (size_t)(kb + 2) * 32, 0, 3);
    }

    const __bf16* bA = &sA[cur][0];
    const __bf16* bW = &sW[cur][0];

    v16bf fa[2];
#pragma unroll
    for (int mi = 0; mi < 2; ++mi) {
      const __bf16* p = bA + (wm * 32 + mi * 16 + fr) * PAD + kh;
      v8bf lo = *(const v8bf*)(p);
      v8bf hi = *(const v8bf*)(p + 16);
      fa[mi] = __builtin_shufflevector(lo, hi, 0,1,2,3,4,5,6,7,8,9,10,11,12,13,14,15);
    }
    v16bf fb[WN];
#pragma unroll
    for (int ni = 0; ni < WN; ++ni) {
      const __bf16* p = bW + (wn * (16 * WN) + ni * 16 + fr) * PAD + kh;
      v8bf lo = *(const v8bf*)(p);
      v8bf hi = *(const v8bf*)(p + 16);
      fb[ni] = __builtin_shufflevector(lo, hi, 0,1,2,3,4,5,6,7,8,9,10,11,12,13,14,15);
    }

#pragma unroll
    for (int mi = 0; mi < 2; ++mi)
#pragma unroll
      for (int ni = 0; ni < WN; ++ni)
        c[mi][ni] = __builtin_amdgcn_wmma_f32_16x16x32_bf16(
            false, fa[mi], false, fb[ni], (short)0, c[mi][ni], false, false);

    __syncthreads();
    if (kb + 1 < nk) {
      const int nxt = cur ^ 1;
#pragma unroll
      for (int i = 0; i < 2; ++i)
        *(v8bf*)&sA[nxt][(lrow + 32 * i) * PAD + lseg] = ra[i];
#pragma unroll
      for (int i = 0; i < WSL; ++i)
        *(v8bf*)&sW[nxt][(lrow + 32 * i) * PAD + lseg] = rw[i];
      __syncthreads();
    }
  }

  // Epilogue. C/D layout: lanes 0-15 -> N=lane, M = vgpr(0..7);
  //                       lanes 16-31 -> N=lane-16, M = vgpr+8.
  const float s    = scale ? scale[0] : 1.0f;
  const int   ncol = lane & 15;
  const int   moff = (lane >> 4) * 8;
#pragma unroll
  for (int mi = 0; mi < 2; ++mi) {
#pragma unroll
    for (int ni = 0; ni < WN; ++ni) {
      const int gm = m0 + wm * 32 + mi * 16 + moff;
      const int gn = n0 + wn * (16 * WN) + ni * 16 + ncol;
      const float bb = bias[gn];
#pragma unroll
      for (int r = 0; r < 8; ++r) {
        const float val = c[mi][ni][r] * s + bb;
        if (outF) outF[(size_t)(gm + r) * ldo + gn] = val;
        if (outB) outB[(size_t)(gm + r) * ldob + gn] = (__bf16)val;
      }
    }
  }
}

// ---------------------------------------------------------------------------
// GRU gate elementwise: h_new = (1-u)*n + u*h  (fp32 state, bf16 copy)
// ---------------------------------------------------------------------------
__global__ void k_gates(const float* __restrict__ gi, const float* __restrict__ gh,
                        float* __restrict__ hf, __bf16* __restrict__ hb) {
  const int idx = blockIdx.x * 256 + threadIdx.x;   // Bsz*Hd threads
  const int b = idx >> 11;                          // / 2048
  const int j = idx & (Hd - 1);
  const float* gib = gi + (size_t)b * G3;
  const float* ghb = gh + (size_t)b * G3;
  float r  = gib[j]          + ghb[j];
  float u  = gib[Hd + j]     + ghb[Hd + j];
  float in = gib[2 * Hd + j];
  float hn = ghb[2 * Hd + j];
  r = 1.f / (1.f + __expf(-r));
  u = 1.f / (1.f + __expf(-u));
  const float n = tanhf(in + r * hn);
  const float h = hf[idx];
  const float hnew = (1.f - u) * n + u * h;
  hf[idx] = hnew;
  hb[idx] = (__bf16)hnew;
}

// ---------------------------------------------------------------------------
// In-place BatchNorm over the batch axis for each (t,z).
// d_out layout: out[(b*T + t)*Z + z].  One thread per (t,z) column.
// ---------------------------------------------------------------------------
__global__ void k_batchnorm(float* __restrict__ out) {
  const int id = blockIdx.x * 256 + threadIdx.x;    // Td*Zd threads
  const int t = id >> 9;                            // / 512
  const int z = id & (Zd - 1);
  float s = 0.f, ss = 0.f;
  for (int b = 0; b < Bsz; ++b) {
    const float v = out[((size_t)b * Td + t) * Zd + z];
    s += v; ss += v * v;
  }
  const float mean = s * (1.f / Bsz);
  const float var  = fmaxf(ss * (1.f / Bsz) - mean * mean, 0.f);
  const float inv  = rsqrtf(var + 1e-5f);
  for (int b = 0; b < Bsz; ++b) {
    const size_t o = ((size_t)b * Td + t) * Zd + z;
    out[o] = (out[o] - mean) * inv;
  }
}

// ---------------------------------------------------------------------------
// Host driver
// ---------------------------------------------------------------------------
extern "C" void kernel_launch(void* const* d_in, const int* in_sizes, int n_in,
                              void* d_out, int out_size, void* d_ws, size_t ws_size,
                              hipStream_t stream) {
  (void)in_sizes; (void)n_in; (void)out_size; (void)ws_size;
  const float* z_in  = (const float*)d_in[0];
  const float* fc_w  = (const float*)d_in[1];
  const float* fc_b  = (const float*)d_in[2];
  const float* fc_u  = (const float*)d_in[3];
  const float* w_ih  = (const float*)d_in[4];
  const float* w_hh  = (const float*)d_in[5];
  const float* b_ih  = (const float*)d_in[6];
  const float* b_hh  = (const float*)d_in[7];
  const float* lin_w = (const float*)d_in[8];
  const float* lin_b = (const float*)d_in[9];
  float* out = (float*)d_out;

  // workspace carve-out (256B aligned slices)
  char* ws = (char*)d_ws;
  size_t off = 0;
  auto carve = [&](size_t bytes) -> char* {
    char* p = ws + off;
    off = (off + bytes + 255) & ~(size_t)255;
    return p;
  };
  float* svt   = (float*)carve(Zd * 4);
  float* svv   = (float*)carve(Zd * 4);
  float* swv   = (float*)carve(Zd * 4);
  float* scal  = (float*)carve(64);
  float* gi    = (float*)carve((size_t)Bsz * G3 * 4);
  float* gh    = (float*)carve((size_t)Bsz * G3 * 4);
  float* hf    = (float*)carve((size_t)Bsz * Hd * 4);
  __bf16* fcwb = (__bf16*)carve((size_t)Zd * Zd * 2);
  __bf16* wihb = (__bf16*)carve((size_t)G3 * Zd * 2);
  __bf16* whhb = (__bf16*)carve((size_t)G3 * Hd * 2);
  __bf16* linwb= (__bf16*)carve((size_t)Zd * Hd * 2);
  __bf16* zb   = (__bf16*)carve((size_t)Bsz * Zd * 2);
  __bf16* xb   = (__bf16*)carve((size_t)Bsz * Zd * 2);
  __bf16* hb   = (__bf16*)carve((size_t)Bsz * Hd * 2);

  // 1) bf16 weight/activation copies (one-time per call; ~36MB, HBM-cheap)
  auto conv = [&](const float* s, __bf16* d, int n) {
    int blocks = (n + 256 * 8 - 1) / (256 * 8);
    if (blocks > 4096) blocks = 4096;
    k_f32_to_bf16<<<blocks, 256, 0, stream>>>(s, d, n);
  };
  conv(z_in,  zb,   Bsz * Zd);
  conv(fc_w,  fcwb, Zd * Zd);
  conv(w_ih,  wihb, G3 * Zd);
  conv(w_hh,  whhb, G3 * Hd);
  conv(lin_w, linwb, Zd * Hd);

  // 2) h0 = 0 (fp32 + bf16 copies)
  k_zero_u32<<<512, 256, 0, stream>>>((uint32_t*)hf, Bsz * Hd);
  k_zero_u32<<<256, 256, 0, stream>>>((uint32_t*)hb, Bsz * Hd / 2);

  // 3) spectral norm -> 1/sigma
  k_wT_u<<<2, 256, 0, stream>>>(fc_w, fc_u, svt);
  k_normalize<<<1, 256, 0, stream>>>(svt, svv);
  k_w_v<<<2, 256, 0, stream>>>(fc_w, svv, swv);
  k_sigma<<<1, 256, 0, stream>>>(swv, scal);

  // 4) x0 = z @ (W/sigma)^T + fc_b  -> bf16 x feed   (narrow config: BN=64)
  k_gemm_bf16<2><<<dim3(Zd / 64, Bsz / 64), 128, 0, stream>>>(
      zb, Zd, fcwb, Zd, fc_b, scal,
      nullptr, 0, xb, Zd, Zd);

  // 5) recurrence
  for (int t = 0; t < Td; ++t) {
    // gi = x @ w_ih^T + b_ih   [256 x 6144], K=512   (wide config: BN=128)
    k_gemm_bf16<4><<<dim3(G3 / 128, Bsz / 64), 128, 0, stream>>>(
        xb, Zd, wihb, Zd, b_ih, nullptr,
        gi, G3, nullptr, 0, Zd);
    // gh = h @ w_hh^T + b_hh   [256 x 6144], K=2048  (wide config)
    k_gemm_bf16<4><<<dim3(G3 / 128, Bsz / 64), 128, 0, stream>>>(
        hb, Hd, whhb, Hd, b_hh, nullptr,
        gh, G3, nullptr, 0, Hd);
    // gates -> h_new (fp32 + bf16)
    k_gates<<<(Bsz * Hd) / 256, 256, 0, stream>>>(gi, gh, hf, hb);
    // x_new = h_new @ lin_w^T + lin_b -> transposed slot of d_out + bf16 feed
    k_gemm_bf16<2><<<dim3(Zd / 64, Bsz / 64), 128, 0, stream>>>(
        hb, Hd, linwb, Hd, lin_b, nullptr,
        out + (size_t)t * Zd, Td * Zd, xb, Zd, Hd);
  }

  // 6) per-(t,z) batchnorm over batch, in place on d_out
  k_batchnorm<<<(Td * Zd) / 256, 256, 0, stream>>>(out);
}
